// WasteReasoningRGN_84791244358307
// MI455X (gfx1250) — compile-verified, matrix operand order checked by reading, not run on hardware
//
#include <hip/hip_runtime.h>
#include <hip/hip_bf16.h>
#include <stdint.h>

typedef __attribute__((ext_vector_type(16))) _Float16 v16h;
typedef __attribute__((ext_vector_type(8)))  _Float16 v8h;
typedef __attribute__((ext_vector_type(8)))  float    v8f;

#define IN_F  128
#define OUT_F 128
#define N_REL 4

// ---------------------------------------------------------------------------
// Kernel 0: convert + transpose all 5 weight matrices to f16 once.
//   Wt[mid*128 + n][k] = (f16) W_mid[k][n]        (Wt: [640, 128] f16, 160 KB)
// One block per (mid, n) output row; thread = k; contiguous f16 writes.
// ---------------------------------------------------------------------------
__global__ __launch_bounds__(128) void rgnn_cvt_weights(
    const float* __restrict__ W_self, const float* __restrict__ W_rel,
    _Float16* __restrict__ Wt)
{
    const int bid = blockIdx.x;        // 0..639
    const int mid = bid >> 7;          // 0..4
    const int n   = bid & 127;
    const int k   = threadIdx.x;       // 0..127
    const float* W = (mid == 0) ? W_self
                                : (W_rel + (size_t)(mid - 1) * IN_F * OUT_F);
    Wt[(size_t)bid * IN_F + k] = (_Float16)W[(size_t)k * OUT_F + n];
}

// ---------------------------------------------------------------------------
// Kernel 1: fused 5-way GEMM with WMMA. One block per 16-row tile of x.
// 8 waves; wave w owns output columns [16w,16w+16). Each wave loads/converts
// its A fragments ONCE, then sweeps mid = 0..4 (W_self + 4 relations):
//   mid 0 -> d_out (stride 128), mid r -> H[:,r-1,:] (stride 512).
// Two accumulators break the WMMA->WMMA dependency chain.
// ---------------------------------------------------------------------------
__global__ __launch_bounds__(256) void rgnn_gemm_wmma(
    const float* __restrict__ x,
    const _Float16* __restrict__ Wt,
    const float* __restrict__ b_rel, const float* __restrict__ b_self,
    float* __restrict__ out, float* __restrict__ H)
{
    const int lane  = threadIdx.x & 31;
    const int wave  = threadIdx.x >> 5;     // n-tile
    const int mbase = blockIdx.x << 4;      // 16 rows
    const int nbase = wave << 4;
    const int m     = lane & 15;            // A row / B column within tile
    const int hi    = lane >> 4;            // K-half selector (wave32 layout)

    // ---- A fragments (4 K-chunks), loaded and converted once ----
    v16h a[4];
    const float* xrow = x + (size_t)(mbase + m) * IN_F;
    #pragma unroll
    for (int kc = 0; kc < 4; ++kc) {
        const int k0 = kc * 32;
        float af[16];
        *(float4*)(af + 0)  = *(const float4*)(xrow + k0 + hi * 8);
        *(float4*)(af + 4)  = *(const float4*)(xrow + k0 + hi * 8 + 4);
        *(float4*)(af + 8)  = *(const float4*)(xrow + k0 + 16 + hi * 8);
        *(float4*)(af + 12) = *(const float4*)(xrow + k0 + 16 + hi * 8 + 4);
        #pragma unroll
        for (int i = 0; i < 16; ++i) a[kc][i] = (_Float16)af[i];
    }

    const int nn   = lane & 15;
    const int mofs = hi << 3;

    #pragma unroll
    for (int mid = 0; mid < 5; ++mid) {
        const _Float16* wrow = Wt + (size_t)(mid * 128 + nbase + m) * IN_F;

        v8f acc0 = {}, acc1 = {};
        #pragma unroll
        for (int kc = 0; kc < 4; ++kc) {
            const int k0 = kc * 32;
            // B fragment: two contiguous 16-byte f16 runs per chunk
            const v8h b0 = *(const v8h*)(wrow + k0 + hi * 8);
            const v8h b1 = *(const v8h*)(wrow + k0 + 16 + hi * 8);
            v16h b;
            #pragma unroll
            for (int i = 0; i < 8; ++i) { b[i] = b0[i]; b[8 + i] = b1[i]; }

            if (kc & 1)
                acc1 = __builtin_amdgcn_wmma_f32_16x16x32_f16(
                    false, a[kc], false, b, (short)0, acc1, false, false);
            else
                acc0 = __builtin_amdgcn_wmma_f32_16x16x32_f16(
                    false, a[kc], false, b, (short)0, acc0, false, false);
        }

        const float* bias = (mid == 0) ? b_self
                                       : (b_rel + (size_t)(mid - 1) * OUT_F);
        float*  base    = (mid == 0) ? out : (H + (size_t)(mid - 1) * OUT_F);
        const size_t rs = (mid == 0) ? (size_t)OUT_F : (size_t)N_REL * OUT_F;

        const float bv = bias[nbase + nn];
        #pragma unroll
        for (int r = 0; r < 8; ++r) {
            const int row = mbase + mofs + r;
            base[(size_t)row * rs + nbase + nn] = acc0[r] + acc1[r] + bv;
        }
    }
}

// ---------------------------------------------------------------------------
// Kernel 2: per-node attention projections (separable attention).
//   a_src[n] = dot(x[n], W_att[0:128]);  a_tgt[n] = dot(x[n], W_att[128:256])
// ---------------------------------------------------------------------------
__global__ __launch_bounds__(256) void rgnn_att_proj(
    const float* __restrict__ x, const float* __restrict__ W_att,
    float* __restrict__ a_src, float* __restrict__ a_tgt, int n_nodes)
{
    const int lane = threadIdx.x & 31;
    const int node = blockIdx.x * 8 + (threadIdx.x >> 5);
    if (node >= n_nodes) return;

    const float4 xv = ((const float4*)(x + (size_t)node * IN_F))[lane];
    const float4 ws = ((const float4*)(W_att))[lane];
    const float4 wt = ((const float4*)(W_att + IN_F))[lane];

    float s = xv.x * ws.x + xv.y * ws.y + xv.z * ws.z + xv.w * ws.w;
    float t = xv.x * wt.x + xv.y * wt.y + xv.z * wt.z + xv.w * wt.w;
    #pragma unroll
    for (int off = 16; off > 0; off >>= 1) {
        s += __shfl_down(s, off, 32);
        t += __shfl_down(t, off, 32);
    }
    if (lane == 0) { a_src[node] = s; a_tgt[node] = t; }
}

// ---------------------------------------------------------------------------
// Kernel 3: edge gather + sigmoid gate + atomic scatter-add.
// One wave per edge: 512B H-row gather (float4/lane), 4 f32 atomics/lane.
// ---------------------------------------------------------------------------
__global__ __launch_bounds__(256) void rgnn_edge_scatter(
    const int* __restrict__ src, const int* __restrict__ tgt,
    const int* __restrict__ rel,
    const float* __restrict__ H,
    const float* __restrict__ a_src, const float* __restrict__ a_tgt,
    const float* __restrict__ b_att,
    float* __restrict__ out, int n_edges)
{
    const int lane = threadIdx.x & 31;
    const int e    = blockIdx.x * 8 + (threadIdx.x >> 5);
    if (e >= n_edges) return;

    const int s = src[e];
    const int t = tgt[e];
    const int r = rel[e];

    const float logit = a_src[s] + a_tgt[t] + b_att[0];
    const float att   = 1.0f / (1.0f + __expf(-logit));

    const float4 m =
        ((const float4*)(H + ((size_t)s * N_REL + r) * OUT_F))[lane];

    float* o = out + (size_t)t * OUT_F + lane * 4;
    atomicAdd(o + 0, m.x * att);
    atomicAdd(o + 1, m.y * att);
    atomicAdd(o + 2, m.z * att);
    atomicAdd(o + 3, m.w * att);
}

// ---------------------------------------------------------------------------
extern "C" void kernel_launch(void* const* d_in, const int* in_sizes, int n_in,
                              void* d_out, int out_size, void* d_ws, size_t ws_size,
                              hipStream_t stream) {
    const float* x      = (const float*)d_in[0];
    const int*   eidx   = (const int*)  d_in[1];   // [2, E]
    const int*   etype  = (const int*)  d_in[2];   // [E]
    const float* W_rel  = (const float*)d_in[3];
    const float* b_rel  = (const float*)d_in[4];
    const float* W_self = (const float*)d_in[5];
    const float* b_self = (const float*)d_in[6];
    const float* W_att  = (const float*)d_in[7];
    const float* b_att  = (const float*)d_in[8];

    const int N = in_sizes[0] / IN_F;   // 50000 (divisible by 16)
    const int E = in_sizes[2];          // 500000

    float* out = (float*)d_out;

    // workspace carve-up: H [N,R,128] f32 | a_src[N] | a_tgt[N] | Wt [640,128] f16
    char* p = (char*)d_ws;
    float* H = (float*)p;       p += (size_t)N * N_REL * OUT_F * sizeof(float);
    float* a_src = (float*)p;   p += (size_t)N * sizeof(float);
    float* a_tgt = (float*)p;   p += (size_t)N * sizeof(float);
    p = (char*)(((uintptr_t)p + 63) & ~(uintptr_t)63);
    _Float16* Wt = (_Float16*)p;

    // 0) weights -> transposed f16 (tiny)
    rgnn_cvt_weights<<<(1 + N_REL) * OUT_F, 128, 0, stream>>>(W_self, W_rel, Wt);

    // 1) fused self + per-relation GEMMs (WMMA), A loaded once per tile
    rgnn_gemm_wmma<<<N / 16, 256, 0, stream>>>(x, Wt, b_rel, b_self, out, H);

    // 2) per-node attention projections
    rgnn_att_proj<<<(N + 7) / 8, 256, 0, stream>>>(x, W_att, a_src, a_tgt, N);

    // 3) edge gather / gate / scatter-add into d_out
    const int* srcp = eidx;
    const int* tgtp = eidx + E;
    rgnn_edge_scatter<<<(E + 7) / 8, 256, 0, stream>>>(srcp, tgtp, etype, H,
                                                       a_src, a_tgt, b_att,
                                                       out, E);
}
